// DINODecoder_60026462929344
// MI455X (gfx1250) — compile-verified, matrix operand order checked by reading, not run on hardware
//
#include <hip/hip_runtime.h>
#include <hip/hip_bf16.h>
#include <math.h>

// ---------------------------------------------------------------------------
// DINO deformable decoder for MI455X (gfx1250).
// - Every dense GEMM runs on v_wmma_f32_16x16x32_bf16 (fp32 accumulate),
//   2x2 register-blocked per wave (32x32 tile) so each A/B fragment feeds
//   two WMMAs (halves bytes/FLOP vs the 1x1 version).
// - Self-attention is WMMA too: head dim 32 == one K-step for Q.K^T; P.V
//   accumulates over 928 (padded) keys with probabilities staged in LDS and
//   V stored transposed so B-fragments are contiguous.
// - memory / value tensors stay bf16 to halve the dominant HBM bytes.
// ---------------------------------------------------------------------------

typedef __attribute__((ext_vector_type(16))) __bf16 v16bf;
typedef __attribute__((ext_vector_type(8)))  __bf16 v8bf;
typedef __attribute__((ext_vector_type(8)))  float  v8f;

#define B_    4
#define NQ_   900
#define LV_   20197
#define NL_   6
#define KPAD  928          // 29 * 32 key padding for P.V WMMA

static __device__ __forceinline__ __bf16 f2bf(float f) {
  unsigned u = __builtin_bit_cast(unsigned, f);
  unsigned r = u + 0x7fffu + ((u >> 16) & 1u);           // round-to-nearest-even
  unsigned short h = (unsigned short)(r >> 16);
  return __builtin_bit_cast(__bf16, h);
}
static __device__ __forceinline__ float bf2f(__bf16 b) {
  unsigned short h = __builtin_bit_cast(unsigned short, b);
  return __builtin_bit_cast(float, ((unsigned)h) << 16);
}
static __device__ __forceinline__ float sigm(float x) { return 1.0f / (1.0f + __expf(-x)); }
static __device__ __forceinline__ v16bf cat16(v8bf a, v8bf b) {
  return __builtin_shufflevector(a, b, 0,1,2,3,4,5,6,7,8,9,10,11,12,13,14,15);
}

// ---------------------------------------------------------------------------
// WMMA GEMM: C[b,m,n] = A[b,m,:] . W[n,:] (+bias[n]) (+resid) (ReLU?)
// A: bf16 [B, M, K]; W: bf16 [N, K] row-major (x @ W^T).
// Grid: (ceil(M/64), N/64, B); block 128 = 4 waves in 2x2; 32x32 tile/wave.
// tstore: write Cb transposed as Cb[b*sC + col*ldt + m] (for V^T).
// ---------------------------------------------------------------------------
__global__ __launch_bounds__(128) void k_gemm(
    const __bf16* __restrict__ A, const __bf16* __restrict__ W,
    const float* __restrict__ bias, const float* __restrict__ resid,
    float* __restrict__ Cf, __bf16* __restrict__ Cb,
    int M, int K, int ldc, long sA, long sC, int relu, int tstore, int ldt)
{
  const int lane = threadIdx.x & 31;
  const int wv   = threadIdx.x >> 5;
  const int wm = wv >> 1, wn = wv & 1;
  const int b  = blockIdx.z;
  const int mBase = blockIdx.x * 64 + wm * 32;
  const int nBase = blockIdx.y * 64 + wn * 32;
  const int ln = lane & 15, kh = lane >> 4;

  const __bf16* Ap[2];
  const __bf16* Wp[2];
#pragma unroll
  for (int i = 0; i < 2; ++i) {
    int ar = mBase + i * 16 + ln; if (ar >= M) ar = M - 1;   // clamp; stores guarded
    Ap[i] = A + (long)b * sA + (long)ar * K + kh * 8;
    Wp[i] = W + (long)(nBase + i * 16 + ln) * K + kh * 8;
  }

  v8f acc[2][2];
#pragma unroll
  for (int mi = 0; mi < 2; ++mi)
#pragma unroll
    for (int ni = 0; ni < 2; ++ni)
      acc[mi][ni] = (v8f){0.f,0.f,0.f,0.f,0.f,0.f,0.f,0.f};

  for (int kb = 0; kb < K; kb += 32) {
    __builtin_prefetch((const void*)(Ap[0] + kb + 128), 0, 1);
    __builtin_prefetch((const void*)(Ap[1] + kb + 128), 0, 1);
    v16bf av[2], bv[2];
#pragma unroll
    for (int i = 0; i < 2; ++i) {
      av[i] = cat16(*(const v8bf*)(Ap[i] + kb), *(const v8bf*)(Ap[i] + kb + 16));
      bv[i] = cat16(*(const v8bf*)(Wp[i] + kb), *(const v8bf*)(Wp[i] + kb + 16));
    }
#pragma unroll
    for (int mi = 0; mi < 2; ++mi)
#pragma unroll
      for (int ni = 0; ni < 2; ++ni)
        acc[mi][ni] = __builtin_amdgcn_wmma_f32_16x16x32_bf16(
            false, av[mi], false, bv[ni], (short)0, acc[mi][ni], false, false);
  }

#pragma unroll
  for (int ni = 0; ni < 2; ++ni) {
    const int col = nBase + ni * 16 + ln;
    const float bb = bias ? bias[col] : 0.0f;
#pragma unroll
    for (int mi = 0; mi < 2; ++mi) {
      const int mB = mBase + mi * 16 + 8 * kh;   // C layout: VGPR r -> M = r (+8 hi lanes)
#pragma unroll
      for (int r = 0; r < 8; ++r) {
        int m = mB + r;
        if (m < M) {
          long ci = (long)b * sC + (long)m * ldc + col;
          float v = acc[mi][ni][r] + bb;
          if (resid) v += resid[ci];
          if (relu)  v = fmaxf(v, 0.0f);
          if (Cf) Cf[ci] = v;
          if (Cb) {
            long co = tstore ? ((long)b * sC + (long)col * ldt + m) : ci;
            Cb[co] = f2bf(v);
          }
        }
      }
    }
  }
}

// f32 -> bf16 convert
__global__ void k_cvt(const float* __restrict__ in, __bf16* __restrict__ out, long n) {
  long i = (long)blockIdx.x * 256 + threadIdx.x;
  if (i < n) out[i] = f2bf(in[i]);
}

// out_bf = bf16(a + b)
__global__ void k_addcvt(const float* __restrict__ a, const float* __restrict__ b,
                         __bf16* __restrict__ out, long n) {
  long i = (long)blockIdx.x * 256 + threadIdx.x;
  if (i < n) out[i] = f2bf(a[i] + b[i]);
}

__global__ void k_sigmoid(const float* __restrict__ in, float* __restrict__ out, long n) {
  long i = (long)blockIdx.x * 256 + threadIdx.x;
  if (i < n) out[i] = sigm(in[i]);
}

// feat [B,256,H,W] f32  ->  mem [B, LV, 256] bf16 (rows = spatial, cols = channel)
__global__ void k_permute(const float* __restrict__ feat, __bf16* __restrict__ mem,
                          int off, int HW) {
  int i = blockIdx.x * 256 + threadIdx.x;
  int b = blockIdx.y;
  if (i >= HW * 256) return;
  int c = i / HW, p = i % HW;                    // coalesced reads along W
  mem[((long)b * LV_ + off + p) * 256 + c] = f2bf(feat[((long)b * 256 + c) * HW + p]);
}

// zero the key-pad columns (900..927) of vT [B,256,KPAD]
__global__ void k_zpad(__bf16* __restrict__ vT) {
  int i = blockIdx.x * 256 + threadIdx.x;        // B*256*28
  if (i >= B_ * 256 * 28) return;
  int j = i % 28, bd = i / 28;
  vT[(long)bd * KPAD + 900 + j] = f2bf(0.f);
}

// DINO gen_sineembed_for_position: ref(sig) [B,NQ,4] -> emb bf16 [B,NQ,512]
__global__ void k_sine(const float* __restrict__ ref, __bf16* __restrict__ emb, long n) {
  long i = (long)blockIdx.x * 256 + threadIdx.x;
  if (i >= n) return;
  int  d  = (int)(i & 511);
  long bq = i >> 9;
  int comp = d >> 7, r = d & 127, j = r >> 1;
  const int map[4] = {1, 0, 2, 3};               // concat order: y, x, w, h
  float v = ref[bq * 4 + map[comp]];
  float t = v * 6.28318530718f * __expf(-(float)j * 0.14391156831f); // /10000^(j/64)
  emb[i] = f2bf((r & 1) ? __cosf(t) : __sinf(t));
}

// ---------------------------------------------------------------------------
// WMMA self-attention. Block = 128 thr per (16-query tile, head, batch).
// qk bf16 [B,NQ,512] (q | k); vT bf16 [B,256,KPAD] (dim-major, keys padded 0).
// Scores -> LDS f32; softmax (scale folded into exp); P.V WMMA with 1/sum
// folded into the epilogue. EXEC stays all-1s around every WMMA.
// ---------------------------------------------------------------------------
__global__ __launch_bounds__(128) void k_attn2(const __bf16* __restrict__ qk,
                                               const __bf16* __restrict__ vT,
                                               __bf16* __restrict__ outb) {
  __shared__ float sc[16][KPAD];
  __shared__ float red[16][8];
  __shared__ float rmax[16];
  __shared__ float rsum[16];
  __shared__ float part[2][16][16];
  const int qt = blockIdx.x, h = blockIdx.y, b = blockIdx.z;
  const int tid = threadIdx.x;
  const int wv = tid >> 5, lane = tid & 31;
  const int ln = lane & 15, kh = lane >> 4;

  // ---- scores: S = Q K^T (K-dim = head dim 32 -> one WMMA per key tile) ----
  int qrow = qt * 16 + ln; if (qrow > NQ_ - 1) qrow = NQ_ - 1;
  const __bf16* Qp = qk + ((long)b * NQ_ + qrow) * 512 + h * 32 + kh * 8;
  v16bf qa = cat16(*(const v8bf*)Qp, *(const v8bf*)(Qp + 16));
  for (int kt = wv; kt < 57; kt += 4) {
    int krow = kt * 16 + ln; if (krow > NQ_ - 1) krow = NQ_ - 1;
    const __bf16* Kp = qk + ((long)b * NQ_ + krow) * 512 + 256 + h * 32 + kh * 8;
    v16bf kv = cat16(*(const v8bf*)Kp, *(const v8bf*)(Kp + 16));
    v8f acc = {0.f,0.f,0.f,0.f,0.f,0.f,0.f,0.f};
    acc = __builtin_amdgcn_wmma_f32_16x16x32_bf16(false, qa, false, kv,
                                                  (short)0, acc, false, false);
    const int col = kt * 16 + ln;
#pragma unroll
    for (int r = 0; r < 8; ++r) sc[r + 8 * kh][col] = acc[r];
  }
  __syncthreads();

  // ---- softmax over keys (scaled by 1/sqrt(32) inside exp) ----
  const int rr = tid & 15, seg = tid >> 4;
  float mx = -1e30f;
  for (int j = seg; j < NQ_; j += 8) mx = fmaxf(mx, sc[rr][j]);
  red[rr][seg] = mx; __syncthreads();
  if (tid < 16) {
    float m = red[tid][0];
    for (int s = 1; s < 8; ++s) m = fmaxf(m, red[tid][s]);
    rmax[tid] = m;
  }
  __syncthreads();
  float m = rmax[rr], lsum = 0.f;
  for (int j = seg; j < NQ_; j += 8) {
    float e = __expf((sc[rr][j] - m) * 0.176776695f);
    sc[rr][j] = e; lsum += e;
  }
  red[rr][seg] = lsum; __syncthreads();
  if (tid < 16) {
    float s = 0.f;
    for (int k = 0; k < 8; ++k) s += red[tid][k];
    rsum[tid] = 1.0f / s;
  }
  for (int idx = tid; idx < 16 * 28; idx += 128)   // zero pad keys 900..927
    sc[idx / 28][NQ_ + idx % 28] = 0.f;
  __syncthreads();

  // ---- O = P V : K = KPAD keys, split 2 ways; dims split 2 ways ----
  const int wn = wv & 1, ws = wv >> 1;
  v8f acc = {0.f,0.f,0.f,0.f,0.f,0.f,0.f,0.f};
  const int dim = h * 32 + wn * 16 + ln;
  const __bf16* Vp = vT + ((long)b * 256 + dim) * KPAD + kh * 8;
  const float* srow = sc[ln];
  for (int kt = ws; kt < KPAD / 32; kt += 2) {
    const int k0 = kt * 32 + kh * 8;
    v16bf pa;
#pragma unroll
    for (int i = 0; i < 8; ++i) {
      pa[i]     = f2bf(srow[k0 + i]);
      pa[8 + i] = f2bf(srow[k0 + 16 + i]);
    }
    v16bf vv = cat16(*(const v8bf*)(Vp + kt * 32), *(const v8bf*)(Vp + kt * 32 + 16));
    acc = __builtin_amdgcn_wmma_f32_16x16x32_bf16(false, pa, false, vv,
                                                  (short)0, acc, false, false);
  }
  if (ws == 1) {
#pragma unroll
    for (int r = 0; r < 8; ++r) part[wn][r + 8 * kh][ln] = acc[r];
  }
  __syncthreads();
  if (ws == 0) {
#pragma unroll
    for (int r = 0; r < 8; ++r) {
      const int mrow = r + 8 * kh;
      const int qr = qt * 16 + mrow;
      if (qr < NQ_) {
        float o = (acc[r] + part[wn][mrow][ln]) * rsum[mrow];
        outb[((long)b * NQ_ + qr) * 256 + h * 32 + wn * 16 + ln] = f2bf(o);
      }
    }
  }
}

// softmax over the 16 (lev,point) weights per (b,q,head); a f32 [B,NQ,128] in-place
__global__ void k_sm16(float* __restrict__ a) {
  const int q = blockIdx.x, b = blockIdx.y, t = threadIdx.x;
  long base = ((long)b * NQ_ + q) * 128;
  float v = a[base + t];
  float m = v;
  for (int o = 8; o > 0; o >>= 1) m = fmaxf(m, __shfl_xor(m, o, 16));
  float e = __expf(v - m);
  float s = e;
  for (int o = 8; o > 0; o >>= 1) s += __shfl_xor(s, o, 16);
  a[base + t] = e / s;
}

// Multi-scale deformable sampling: value bf16 [B,LV,256], samp f32 [B,NQ,256],
// attw f32 [B,NQ,128] (softmaxed), ref f32 [B,NQ,4] -> out bf16 [B,NQ,256]
__global__ void k_deform(const __bf16* __restrict__ val, const float* __restrict__ samp,
                         const float* __restrict__ attw, const float* __restrict__ ref,
                         __bf16* __restrict__ outb) {
  const int q = blockIdx.x, b = blockIdx.y;
  const int t = threadIdx.x;
  const int h = t >> 5, d = t & 31;
  const long bq = (long)b * NQ_ + q;
  const float cx = ref[bq * 4 + 0], cy = ref[bq * 4 + 1];
  const float cw = ref[bq * 4 + 2], ch = ref[bq * 4 + 3];
  const int Hs[4]  = {13, 25, 50, 100};
  const int Ws[4]  = {19, 38, 76, 152};
  const int Off[4] = {0, 247, 1197, 4997};
  const float* sampq = samp + bq * 256;
  const float* attq  = attw + bq * 128;
  float acc = 0.0f;
  for (int lev = 0; lev < 4; ++lev) {
    const int Hl = Hs[lev], Wl = Ws[lev];
    const __bf16* vb = val + ((long)b * LV_ + Off[lev]) * 256 + h * 32 + d;
    for (int p = 0; p < 4; ++p) {
      const int oi = (((h * 4 + lev) * 4) + p) * 2;
      const float lx = cx + sampq[oi]     * 0.25f * cw * 0.5f;  // off/NP * wh * 0.5
      const float ly = cy + sampq[oi + 1] * 0.25f * ch * 0.5f;
      const float wgt = attq[h * 16 + lev * 4 + p];
      float x = lx * Wl - 0.5f, y = ly * Hl - 0.5f;
      float xf = floorf(x), yf = floorf(y);
      int x0 = (int)xf, y0 = (int)yf;
      float wx = x - xf, wy = y - yf;
      float s = 0.0f;
#pragma unroll
      for (int dy = 0; dy < 2; ++dy)
#pragma unroll
        for (int dx = 0; dx < 2; ++dx) {
          int xx = x0 + dx, yy = y0 + dy;
          float ww = (dy ? wy : 1.f - wy) * (dx ? wx : 1.f - wx);
          if (xx >= 0 && xx < Wl && yy >= 0 && yy < Hl)
            s += ww * bf2f(vb[((long)yy * Wl + xx) * 256]);
        }
      acc += wgt * s;
    }
  }
  outb[bq * 256 + t] = f2bf(acc);
}

// LayerNorm over D=256; one block (256 thr) per row; optional f32 + bf16 outputs
__global__ void k_ln(const float* __restrict__ in, const float* __restrict__ w,
                     const float* __restrict__ bias,
                     float* __restrict__ outf, __bf16* __restrict__ outb) {
  __shared__ float red[256];
  const long row = blockIdx.x;
  const int t = threadIdx.x;
  float v = in[row * 256 + t];
  red[t] = v; __syncthreads();
  for (int s = 128; s > 0; s >>= 1) { if (t < s) red[t] += red[t + s]; __syncthreads(); }
  const float mu = red[0] * (1.0f / 256.0f);
  __syncthreads();
  float d = v - mu;
  red[t] = d * d; __syncthreads();
  for (int s = 128; s > 0; s >>= 1) { if (t < s) red[t] += red[t + s]; __syncthreads(); }
  const float var = red[0] * (1.0f / 256.0f);
  float o = d * rsqrtf(var + 1e-5f) * w[t] + bias[t];
  if (outf) outf[row * 256 + t] = o;
  if (outb) outb[row * 256 + t] = f2bf(o);
}

// tiny N=4 output head (bb_w3): one thread per (b,q,n)
__global__ void k_gemv4(const __bf16* __restrict__ A, const __bf16* __restrict__ W,
                        const float* __restrict__ bias, float* __restrict__ out, long n) {
  long i = (long)blockIdx.x * 128 + threadIdx.x;
  if (i >= n) return;
  int nn = (int)(i & 3);
  long row = i >> 2;
  float acc = bias[nn];
  for (int k = 0; k < 256; ++k) acc += bf2f(A[row * 256 + k]) * bf2f(W[nn * 256 + k]);
  out[i] = acc;
}

// out = sigmoid(delta + inverse_sigmoid(refin))
__global__ void k_boxout(const float* __restrict__ refin, const float* __restrict__ delta,
                         float* __restrict__ outp, long n) {
  long i = (long)blockIdx.x * 256 + threadIdx.x;
  if (i >= n) return;
  float x = fminf(fmaxf(refin[i], 0.f), 1.f);
  float inv = __logf(fmaxf(x, 1e-3f) / fmaxf(1.f - x, 1e-3f));
  outp[i] = sigm(delta[i] + inv);
}

// ---------------------------------------------------------------------------
static inline void gemm(hipStream_t st, const __bf16* A, const __bf16* W,
                        const float* bias, const float* resid,
                        float* Cf, __bf16* Cb,
                        int M, int N, int K, int ldc, long sA, long sC, int relu,
                        int tstore = 0, int ldt = 0) {
  dim3 g((M + 63) / 64, N / 64, B_);
  k_gemm<<<g, 128, 0, st>>>(A, W, bias, resid, Cf, Cb, M, K, ldc, sA, sC,
                            relu, tstore, ldt);
}

extern "C" void kernel_launch(void* const* d_in, const int* in_sizes, int n_in,
                              void* d_out, int out_size, void* d_ws, size_t ws_size,
                              hipStream_t stream) {
  (void)in_sizes; (void)n_in; (void)out_size; (void)ws_size;
  const float* fsrc[4] = {(const float*)d_in[3], (const float*)d_in[2],
                          (const float*)d_in[1], (const float*)d_in[0]};  // small -> large
  const int lHW[4]  = {13 * 19, 25 * 38, 50 * 76, 100 * 152};
  const int lOff[4] = {0, 247, 1197, 4997};
  const float* target = (const float*)d_in[4];
  const float* refuns = (const float*)d_in[5];
  const float* prm    = (const float*)d_in[6];
  float* out = (float*)d_out;

  // ---- params offsets (assumed: jax tree_flatten = alphabetical key order) ----
  long cur = 0;
  auto take = [&](long n) { long o = cur; cur += n; return o; };
  const long o_bb_b1 = take(256), o_bb_b2 = take(256), o_bb_b3 = take(4);
  const long o_bb_w1 = take(65536), o_bb_w2 = take(65536), o_bb_w3 = take(1024);
  const long o_dn_b = take(256), o_dn_w = take(256);
  const long o_attn_b = take(6L * 128),  o_attn_w = take(6L * 128 * 256);
  const long o_ffn_b1 = take(6L * 2048), o_ffn_b2 = take(6L * 256);
  const long o_ffn_w1 = take(6L * 2048 * 256), o_ffn_w2 = take(6L * 256 * 2048);
  const long o_n1_b = take(6L * 256), o_n1_w = take(6L * 256);
  const long o_n2_b = take(6L * 256), o_n2_w = take(6L * 256);
  const long o_n3_b = take(6L * 256), o_n3_w = take(6L * 256);
  const long o_out_b = take(6L * 256), o_out_w = take(6L * 65536);
  const long o_sain_b = take(6L * 768), o_sain_w = take(6L * 768 * 256);
  const long o_saout_b = take(6L * 256), o_saout_w = take(6L * 65536);
  const long o_samp_b = take(6L * 256), o_samp_w = take(6L * 65536);
  const long o_val_b = take(6L * 256), o_val_w = take(6L * 65536);
  const long o_rp_b1 = take(256), o_rp_b2 = take(256);
  const long o_rp_w1 = take(256L * 512), o_rp_w2 = take(65536);
  const long PTOT = cur;

  // ---- workspace bump allocator ----
  char* wsp = (char*)d_ws;
  size_t wcur = 0;
  auto alloc = [&](size_t bytes) -> void* {
    void* p = wsp + wcur; wcur += (bytes + 255) & ~(size_t)255; return p;
  };
  const long NX = (long)B_ * NQ_ * 256;   // 921600
  const long N4 = (long)B_ * NQ_ * 4;     // 14400
  __bf16* pbf     = (__bf16*)alloc(PTOT * 2);                 // bf16 mirror of params
  __bf16* mem_bf  = (__bf16*)alloc((long)B_ * LV_ * 256 * 2);
  __bf16* value_bf= (__bf16*)alloc((long)B_ * LV_ * 256 * 2);
  float*  x_f     = (float*) alloc(NX * 4);
  __bf16* x_bf    = (__bf16*)alloc(NX * 2);
  float*  qpos_f  = (float*) alloc(NX * 4);
  __bf16* q_bf    = (__bf16*)alloc(NX * 2);
  __bf16* emb_bf  = (__bf16*)alloc((long)B_ * NQ_ * 512 * 2);
  __bf16* rph_bf  = (__bf16*)alloc(NX * 2);
  __bf16* qk_bf   = (__bf16*)alloc((long)B_ * NQ_ * 512 * 2);
  __bf16* vT_bf   = (__bf16*)alloc((long)B_ * 256 * KPAD * 2);
  __bf16* ao_bf   = (__bf16*)alloc(NX * 2);
  float*  res_f   = (float*) alloc(NX * 4);
  float*  samp_f  = (float*) alloc(NX * 4);
  float*  attw_f  = (float*) alloc((long)B_ * NQ_ * 128 * 4);
  __bf16* mso_bf  = (__bf16*)alloc(NX * 2);
  __bf16* ffh_bf  = (__bf16*)alloc((long)B_ * NQ_ * 2048 * 2);
  __bf16* t1_bf   = (__bf16*)alloc(NX * 2);
  __bf16* t2_bf   = (__bf16*)alloc(NX * 2);
  float*  delta_f = (float*) alloc(N4 * 4);
  float*  ref_f   = (float*) alloc(N4 * 4);
  float*  ref5_f  = (float*) alloc(N4 * 4);
  __bf16* hs_bf   = (__bf16*)alloc(NX * 2);

  auto g1 = [](long n, int bs) { return dim3((unsigned)((n + bs - 1) / bs)); };

  // ---- one-time prep ----
  k_cvt<<<g1(PTOT, 256), 256, 0, stream>>>(prm, pbf, PTOT);
  for (int lev = 0; lev < 4; ++lev)
    k_permute<<<dim3((lHW[lev] * 256 + 255) / 256, B_), 256, 0, stream>>>(
        fsrc[lev], mem_bf, lOff[lev], lHW[lev]);
  hipMemcpyAsync(x_f, target, NX * 4, hipMemcpyDeviceToDevice, stream);
  k_cvt<<<g1(NX, 256), 256, 0, stream>>>(target, x_bf, NX);
  k_sigmoid<<<g1(N4, 256), 256, 0, stream>>>(refuns, ref_f, N4);

  const long sX = (long)NQ_ * 256;
  for (int l = 0; l < NL_; ++l) {
    const __bf16* w_sain  = pbf + o_sain_w  + (long)l * 768 * 256;
    const __bf16* w_saout = pbf + o_saout_w + (long)l * 65536;
    const __bf16* w_samp  = pbf + o_samp_w  + (long)l * 65536;
    const __bf16* w_attn  = pbf + o_attn_w  + (long)l * 128 * 256;
    const __bf16* w_val   = pbf + o_val_w   + (long)l * 65536;
    const __bf16* w_out   = pbf + o_out_w   + (long)l * 65536;
    const __bf16* w_ffn1  = pbf + o_ffn_w1  + (long)l * 2048 * 256;
    const __bf16* w_ffn2  = pbf + o_ffn_w2  + (long)l * 256 * 2048;
    const float* b_sain  = prm + o_sain_b  + (long)l * 768;
    const float* b_saout = prm + o_saout_b + (long)l * 256;
    const float* b_samp  = prm + o_samp_b  + (long)l * 256;
    const float* b_attn  = prm + o_attn_b  + (long)l * 128;
    const float* b_val   = prm + o_val_b   + (long)l * 256;
    const float* b_out   = prm + o_out_b   + (long)l * 256;
    const float* b_ffn1  = prm + o_ffn_b1  + (long)l * 2048;
    const float* b_ffn2  = prm + o_ffn_b2  + (long)l * 256;

    // query positional embedding: sine(ref) -> MLP(512->256 relu ->256)
    k_sine<<<g1((long)B_ * NQ_ * 512, 256), 256, 0, stream>>>(ref_f, emb_bf,
                                                             (long)B_ * NQ_ * 512);
    gemm(stream, emb_bf, pbf + o_rp_w1, prm + o_rp_b1, nullptr, nullptr, rph_bf,
         NQ_, 256, 512, 256, (long)NQ_ * 512, sX, 1);
    gemm(stream, rph_bf, pbf + o_rp_w2, prm + o_rp_b2, nullptr, qpos_f, nullptr,
         NQ_, 256, 256, 256, sX, sX, 0);

    // self attention (WMMA): q=k=x+qpos, v=x; V stored transposed
    k_addcvt<<<g1(NX, 256), 256, 0, stream>>>(x_f, qpos_f, q_bf, NX);
    gemm(stream, q_bf, w_sain, b_sain, nullptr, nullptr, qk_bf,
         NQ_, 512, 256, 512, sX, (long)NQ_ * 512, 0);
    gemm(stream, x_bf, w_sain + 512 * 256, b_sain + 512, nullptr, nullptr, vT_bf,
         NQ_, 256, 256, 256, sX, (long)256 * KPAD, 0, /*tstore=*/1, /*ldt=*/KPAD);
    k_zpad<<<g1((long)B_ * 256 * 28, 256), 256, 0, stream>>>(vT_bf);
    k_attn2<<<dim3(57, 8, B_), 128, 0, stream>>>(qk_bf, vT_bf, ao_bf);
    gemm(stream, ao_bf, w_saout, b_saout, x_f, res_f, nullptr,
         NQ_, 256, 256, 256, sX, sX, 0);
    k_ln<<<dim3(B_ * NQ_), 256, 0, stream>>>(res_f, prm + o_n2_w + l * 256,
                                             prm + o_n2_b + l * 256, x_f, x_bf);

    // multi-scale deformable cross attention
    k_addcvt<<<g1(NX, 256), 256, 0, stream>>>(x_f, qpos_f, q_bf, NX);
    gemm(stream, mem_bf, w_val, b_val, nullptr, nullptr, value_bf,
         LV_, 256, 256, 256, (long)LV_ * 256, (long)LV_ * 256, 0);
    gemm(stream, q_bf, w_samp, b_samp, nullptr, samp_f, nullptr,
         NQ_, 256, 256, 256, sX, sX, 0);
    gemm(stream, q_bf, w_attn, b_attn, nullptr, attw_f, nullptr,
         NQ_, 128, 256, 128, sX, (long)NQ_ * 128, 0);
    k_sm16<<<dim3(NQ_, B_), 128, 0, stream>>>(attw_f);
    k_deform<<<dim3(NQ_, B_), 256, 0, stream>>>(value_bf, samp_f, attw_f, ref_f, mso_bf);
    gemm(stream, mso_bf, w_out, b_out, x_f, res_f, nullptr,
         NQ_, 256, 256, 256, sX, sX, 0);
    k_ln<<<dim3(B_ * NQ_), 256, 0, stream>>>(res_f, prm + o_n1_w + l * 256,
                                             prm + o_n1_b + l * 256, x_f, x_bf);

    // FFN
    gemm(stream, x_bf, w_ffn1, b_ffn1, nullptr, nullptr, ffh_bf,
         NQ_, 2048, 256, 2048, sX, (long)NQ_ * 2048, 1);
    gemm(stream, ffh_bf, w_ffn2, b_ffn2, x_f, res_f, nullptr,
         NQ_, 256, 2048, 256, (long)NQ_ * 2048, sX, 0);
    k_ln<<<dim3(B_ * NQ_), 256, 0, stream>>>(res_f, prm + o_n3_w + l * 256,
                                             prm + o_n3_b + l * 256, x_f, x_bf);

    // iterative bbox refinement
    gemm(stream, x_bf, pbf + o_bb_w1, prm + o_bb_b1, nullptr, nullptr, t1_bf,
         NQ_, 256, 256, 256, sX, sX, 1);
    gemm(stream, t1_bf, pbf + o_bb_w2, prm + o_bb_b2, nullptr, nullptr, t2_bf,
         NQ_, 256, 256, 256, sX, sX, 1);
    k_gemv4<<<g1(N4, 128), 128, 0, stream>>>(t2_bf, pbf + o_bb_w3, prm + o_bb_b3,
                                             delta_f, N4);
    if (l == NL_ - 1)
      hipMemcpyAsync(ref5_f, ref_f, N4 * 4, hipMemcpyDeviceToDevice, stream);
    k_boxout<<<g1(N4, 256), 256, 0, stream>>>(ref_f, delta_f, ref_f, N4);
  }

  // ---- outputs: pred_logit [B,NQ,256] | pred_box [B,NQ,4] | hs[-1] [B,NQ,256] ----
  float* out_logit = out;
  float* out_box   = out + NX;
  float* out_hs    = out + NX + N4;
  k_ln<<<dim3(B_ * NQ_), 256, 0, stream>>>(x_f, prm + o_dn_w, prm + o_dn_b, out_hs, hs_bf);
  k_ln<<<dim3(B_ * NQ_), 256, 0, stream>>>(out_hs, prm + o_dn_w, prm + o_dn_b,
                                           out_logit, nullptr);
  gemm(stream, hs_bf, pbf + o_bb_w1, prm + o_bb_b1, nullptr, nullptr, t1_bf,
       NQ_, 256, 256, 256, sX, sX, 1);
  gemm(stream, t1_bf, pbf + o_bb_w2, prm + o_bb_b2, nullptr, nullptr, t2_bf,
       NQ_, 256, 256, 256, sX, sX, 1);
  k_gemv4<<<g1(N4, 128), 128, 0, stream>>>(t2_bf, pbf + o_bb_w3, prm + o_bb_b3, delta_f, N4);
  k_boxout<<<g1(N4, 256), 256, 0, stream>>>(ref5_f, delta_f, out_box, N4);
}
